// PCSScanMamba_35588099015167
// MI455X (gfx1250) — matrix operand order repaired and checked by reading, not compiled, not run on hardware
//
#include <hip/hip_runtime.h>

#define DIMX   512
#define NTOK   49
#define BWIN   512
#define DHALF  256
#define DSTATE 8
#define DTRANK 32
#define EDIM   48
#define NDIR   4
#define BALL   (NDIR * BWIN)      // 2048
#define MPRE   (BWIN * NTOK)      // 25088  (divisible by 16)
#define MALL   (BALL * NTOK)      // 100352 (divisible by 16)

typedef __bf16 bf16;
typedef bf16  bf16x8  __attribute__((ext_vector_type(8)));
typedef bf16  bf16x16 __attribute__((ext_vector_type(16)));
typedef float floatx8 __attribute__((ext_vector_type(8)));
typedef unsigned int uint32x4 __attribute__((ext_vector_type(4)));
typedef int   int32x4  __attribute__((ext_vector_type(4)));
typedef int   int32x8  __attribute__((ext_vector_type(8)));

// ---------------------------------------------------------------------------
// Compile-time token permutations (WIN = 7): h=identity, v=transpose,
// d=diagonal zigzag, a=anti-diagonal zigzag.  We store the INVERSE map:
// GEMM-result row l_src is scattered to token slot inv[dir][l_src].
// ---------------------------------------------------------------------------
struct PermTab { int inv[NDIR][NTOK]; };
static constexpr PermTab make_perms() {
    PermTab p{};
    int fwd[NDIR][NTOK] = {};
    for (int j = 0; j < NTOK; j++) fwd[0][j] = j;                    // h
    for (int j = 0; j < NTOK; j++) fwd[1][j] = (j % 7) * 7 + j / 7;  // v (transpose)
    int c = 0;
    for (int s = 0; s < 13; s++) {
        int rlo = (s - 6 > 0) ? (s - 6) : 0;
        int rhi = (s < 6) ? s : 6;
        for (int r = rlo; r <= rhi; r++) {
            fwd[2][c] = r * 7 + (s - r);          // diag
            fwd[3][c] = r * 7 + (6 - (s - r));    // anti-diag
            c++;
        }
    }
    for (int d = 0; d < NDIR; d++)
        for (int j = 0; j < NTOK; j++) p.inv[d][fwd[d][j]] = j;
    return p;
}
__constant__ PermTab c_perm = make_perms();

// ---------------------------------------------------------------------------
// WMMA inner loop, 16(M) x 64(N) per wave: one A fragment feeds 4 WMMAs.
// A: row-major [M][K] bf16, W: row-major [N][K] (computing A @ W^T).
// Per-lane fragment addressing per CDNA5 ISA 7.12.2:
//   A 16x32 16-bit: lo-half lanes K = 0..7 & 16..23 ; hi-half K = 8..15 & 24..31
//   B 32x16 16-bit: lo-half lanes K = 0..7 & 16..23 ; hi-half K = 8..15 & 24..31
// ---------------------------------------------------------------------------
__device__ __forceinline__ void
wmma_tile4(const bf16* __restrict__ Arow, const bf16* __restrict__ Wrow0,
           int K, int lane, floatx8 acc[4])
{
    const int half = lane >> 4;
    union Frag { bf16x16 v; bf16x8 h[2]; };
#pragma unroll 4
    for (int k0 = 0; k0 < K; k0 += 32) {
        __builtin_prefetch(Arow + k0 + 256, 0, 3);   // global_prefetch_b8, ~8 iters ahead
        __builtin_prefetch(Wrow0 + k0 + 256, 0, 3);
        Frag a;
        a.h[0] = *(const bf16x8*)(Arow + k0 + half * 8);
        a.h[1] = *(const bf16x8*)(Arow + k0 + 16 + half * 8);
#pragma unroll
        for (int j = 0; j < 4; j++) {
            const bf16* wr = Wrow0 + (size_t)j * 16 * K + k0 + half * 8;
            Frag b;
            b.h[0] = *(const bf16x8*)(wr);
            b.h[1] = *(const bf16x8*)(wr + 16);
            acc[j] = __builtin_amdgcn_wmma_f32_16x16x32_bf16(
                false, a.v, false, b.v, (short)0, acc[j], false, false);
        }
    }
}

// ---------------------------------------------------------------------------
// Conversion / weight-prep kernels
// ---------------------------------------------------------------------------
__global__ void k_f32_to_bf16(const float* __restrict__ src, bf16* __restrict__ dst, int n) {
    int i = blockIdx.x * blockDim.x + threadIdx.x;
    if (i < n) dst[i] = (bf16)src[i];
}

__device__ __forceinline__ void gate_softmax(const float* gl, float g[4]) {
    float m = gl[0];
    for (int i = 1; i < 4; i++) m = fmaxf(m, gl[i]);
    float s = 0.f;
    for (int i = 0; i < 4; i++) { g[i] = expf(gl[i] - m); s += g[i]; }
    float inv = 1.f / s;
    for (int i = 0; i < 4; i++) g[i] *= inv;
}

__global__ void k_scale_post_w(const float* __restrict__ w, const float* __restrict__ gl,
                               int dir, bf16* __restrict__ dst) {
    int i = blockIdx.x * blockDim.x + threadIdx.x;
    if (i >= DIMX * DIMX) return;
    float g[4]; gate_softmax(gl, g);
    dst[i] = (bf16)(g[dir] * w[i]);
}

__global__ void k_comb_bias(const float* b0, const float* b1, const float* b2, const float* b3,
                            const float* __restrict__ gl, float* __restrict__ out) {
    int n = blockIdx.x * blockDim.x + threadIdx.x;
    if (n >= DIMX) return;
    float g[4]; gate_softmax(gl, g);
    out[n] = g[0] * b0[n] + g[1] * b1[n] + g[2] * b2[n] + g[3] * b3[n];
}

// ---------------------------------------------------------------------------
// Pre-projections: 4 directions, bias add, inverse-permuted scatter to r_all.
// grid = (MPRE/16/8, DIMX/64, 4), block = 256 (8 waves, 16x64 tile each)
// ---------------------------------------------------------------------------
__global__ void k_gemm_pre(const bf16* __restrict__ A, const bf16* __restrict__ Wall,
                           const float* __restrict__ bh, const float* __restrict__ bv,
                           const float* __restrict__ bd, const float* __restrict__ ba,
                           bf16* __restrict__ r_all)
{
    const int lane = threadIdx.x & 31;
    const int wave = threadIdx.x >> 5;
    const int dir  = blockIdx.z;
    const int m_tile = (blockIdx.x * (blockDim.x >> 5) + wave) * 16;
    const int n0     = blockIdx.y * 64;
    if (m_tile >= MPRE) return;

    const bf16* W = Wall + (size_t)dir * DIMX * DIMX;
    const int rowA = m_tile + (lane & 15);
    floatx8 acc[4] = {};
    wmma_tile4(A + (size_t)rowA * DIMX,
               W + (size_t)(n0 + (lane & 15)) * DIMX, DIMX, lane, acc);

    const float* bias = (dir == 0) ? bh : (dir == 1) ? bv : (dir == 2) ? bd : ba;
    const int half = lane >> 4;
#pragma unroll
    for (int j = 0; j < 4; j++) {
        const int n = n0 + j * 16 + (lane & 15);
        const float bn = bias[n];
#pragma unroll
        for (int r = 0; r < 8; r++) {
            int m = m_tile + r + 8 * half;
            int w = m / NTOK, l = m % NTOK;
            int orow = (dir * BWIN + w) * NTOK + c_perm.inv[dir][l];
            r_all[(size_t)orow * DIMX + n] = (bf16)(acc[j][r] + bn);
        }
    }
}

// ---------------------------------------------------------------------------
// Generic GEMM: C(M,512) = A(M,512) @ W(512,512)^T, bf16 in / bf16 out.
// grid = (M/16/8, DIMX/64), block = 256
// ---------------------------------------------------------------------------
__global__ void k_gemm_bf16(const bf16* __restrict__ A, const bf16* __restrict__ W,
                            bf16* __restrict__ C, int M)
{
    const int lane = threadIdx.x & 31;
    const int wave = threadIdx.x >> 5;
    const int m_tile = (blockIdx.x * (blockDim.x >> 5) + wave) * 16;
    const int n0     = blockIdx.y * 64;
    if (m_tile >= M) return;

    const int rowA = m_tile + (lane & 15);
    floatx8 acc[4] = {};
    wmma_tile4(A + (size_t)rowA * DIMX,
               W + (size_t)(n0 + (lane & 15)) * DIMX, DIMX, lane, acc);

    const int half = lane >> 4;
#pragma unroll
    for (int j = 0; j < 4; j++) {
        const int n = n0 + j * 16 + (lane & 15);
#pragma unroll
        for (int r = 0; r < 8; r++) {
            int m = m_tile + r + 8 * half;
            C[(size_t)m * DIMX + n] = (bf16)acc[j][r];
        }
    }
}

// ---------------------------------------------------------------------------
// Fused post-projections + gating: out = sum_dir (y_dir @ (g_dir*W_dir)^T) + comb_bias
// ---------------------------------------------------------------------------
__global__ void k_gemm_post(const bf16* __restrict__ y_all, const bf16* __restrict__ Wpost,
                            const float* __restrict__ comb_bias, float* __restrict__ out)
{
    const int lane = threadIdx.x & 31;
    const int wave = threadIdx.x >> 5;
    const int m_tile = (blockIdx.x * (blockDim.x >> 5) + wave) * 16;
    const int n0     = blockIdx.y * 64;
    if (m_tile >= MPRE) return;

    const int rowm = m_tile + (lane & 15);
    const int w = rowm / NTOK, l = rowm % NTOK;
    floatx8 acc[4] = {};
    for (int dir = 0; dir < NDIR; dir++) {
        const bf16* Arow = y_all + ((size_t)((dir * BWIN + w) * NTOK) + l) * DIMX;
        const bf16* Wrow = Wpost + (size_t)dir * DIMX * DIMX
                                 + (size_t)(n0 + (lane & 15)) * DIMX;
        wmma_tile4(Arow, Wrow, DIMX, lane, acc);
    }
    const int half = lane >> 4;
#pragma unroll
    for (int j = 0; j < 4; j++) {
        const int n = n0 + j * 16 + (lane & 15);
        const float bn = comb_bias[n];
#pragma unroll
        for (int r = 0; r < 8; r++) {
            int m = m_tile + r + 8 * half;
            out[(size_t)m * DIMX + n] = acc[j][r] + bn;
        }
    }
}

// ---------------------------------------------------------------------------
// Depthwise conv (k=3, SAME, cross-correlation) + SiLU on both halves of xz.
// x half -> x_conv (f32, layout (b,l,256)); z half -> y_concat[..., 256+d] (bf16)
// ---------------------------------------------------------------------------
__global__ void k_conv_silu(const bf16* __restrict__ xz,
                            const float* __restrict__ cxw, const float* __restrict__ cxb,
                            const float* __restrict__ czw, const float* __restrict__ czb,
                            float* __restrict__ x_conv, bf16* __restrict__ y_concat)
{
    size_t i = (size_t)blockIdx.x * blockDim.x + threadIdx.x;
    const size_t total = (size_t)BALL * NTOK * DIMX;
    if (i >= total) return;
    int n = (int)(i % DIMX);
    int l = (int)((i / DIMX) % NTOK);
    size_t b = i / ((size_t)DIMX * NTOK);

    bool isx = (n < DHALF);
    int d = isx ? n : n - DHALF;
    const float* wr = (isx ? cxw : czw) + d * 3;
    float v = isx ? cxb[d] : czb[d];
    const bf16* base = xz + ((size_t)b * NTOK + l) * DIMX + n;
    if (l > 0)        v += wr[0] * (float)base[-DIMX];
    v += wr[1] * (float)base[0];
    if (l < NTOK - 1) v += wr[2] * (float)base[DIMX];
    float s = v / (1.f + expf(-v));          // silu

    if (isx) x_conv[((size_t)b * NTOK + l) * DHALF + d] = s;
    else     y_concat[((size_t)b * NTOK + l) * DIMX + DHALF + d] = (bf16)s;
}

// ---------------------------------------------------------------------------
// x_proj (E=48 dot-256) + LayerNorm over E.  One 64-thread block per (b,l).
// ---------------------------------------------------------------------------
__global__ void k_xproj_ln(const float* __restrict__ x_conv, const float* __restrict__ xpw,
                           const float* __restrict__ lnw, const float* __restrict__ lnb,
                           float* __restrict__ x_dbl)
{
    __shared__ float xs[DHALF];
    __shared__ float es[EDIM];
    __shared__ float red[2];
    const int bl = blockIdx.x;                   // 0 .. MALL-1
    const float* xrow = x_conv + (size_t)bl * DHALF;
    for (int i = threadIdx.x; i < DHALF; i += blockDim.x) xs[i] = xrow[i];
    __syncthreads();

    const int e = threadIdx.x;
    if (e < EDIM) {
        const float* wr = xpw + e * DHALF;
        float s = 0.f;
#pragma unroll 8
        for (int d = 0; d < DHALF; d++) s += xs[d] * wr[d];
        es[e] = s;
    }
    __syncthreads();
    if (threadIdx.x == 0) {
        float m = 0.f;
        for (int j = 0; j < EDIM; j++) m += es[j];
        m /= EDIM;
        float v = 0.f;
        for (int j = 0; j < EDIM; j++) { float t = es[j] - m; v += t * t; }
        v /= EDIM;
        red[0] = m; red[1] = rsqrtf(v + 1e-5f);
    }
    __syncthreads();
    if (e < EDIM)
        x_dbl[(size_t)bl * EDIM + e] = (es[e] - red[0]) * red[1] * lnw[e] + lnb[e];
}

// ---------------------------------------------------------------------------
// Selective scan: block = 256 threads (one d each), one block per batch b.
// x_dbl[b] (49x48 f32 = 9408 B, contiguous) staged into LDS by the Tensor
// Data Mover (1-D D#: data_size=4B, tile_dim0=2352), completion via
// s_wait_tensorcnt + workgroup barrier.  Falls back to cooperative loads
// if the TDM builtin is unavailable.
// ---------------------------------------------------------------------------
__global__ void k_scan(const float* __restrict__ x_dbl, const float* __restrict__ x_conv,
                       const float* __restrict__ dtw, const float* __restrict__ dtb,
                       const float* __restrict__ A_log, const float* __restrict__ Dp,
                       bf16* __restrict__ y_concat)
{
    __shared__ float sdbl[NTOK * EDIM];          // 9408 B
    const int b = blockIdx.x;
    const int d = threadIdx.x;
    const float* src = x_dbl + (size_t)b * NTOK * EDIM;

#if __has_builtin(__builtin_amdgcn_tensor_load_to_lds)
    if (threadIdx.x < 32) {                       // wave 0 drives the TDM
        const unsigned ne = NTOK * EDIM;          // 2352 elements (f32)
        unsigned long long ga = (unsigned long long)(uintptr_t)src;
        unsigned lds_off = (unsigned)(uintptr_t)&sdbl[0];   // low 32b of flat LDS addr
        uint32x4 g0;
        g0[0] = 1u;                               // count=1, is_restore=0, gather off
        g0[1] = lds_off;                          // D#.lds_addr
        g0[2] = (unsigned)(ga & 0xFFFFFFFFu);     // global_addr[31:0]
        g0[3] = (unsigned)((ga >> 32) & 0x1FFFFFFu) | (2u << 30);  // addr[56:32] | type=2
        int32x8 g1;
        g1[0] = (int)(2u << 16);                  // data_size=4B; mask/flags/pad = 0
        g1[1] = (int)((ne & 0xFFFFu) << 16);      // tensor_dim0[15:0] in bits 63:48
        g1[2] = (int)(((ne >> 16) & 0xFFFFu) | (1u << 16)); // tensor_dim0 hi | tensor_dim1=1
        g1[3] = (int)((ne & 0xFFFFu) << 16);      // tensor_dim1 hi (0) | tile_dim0=ne
        g1[4] = 0;                                // tile_dim1=0 (unused), tile_dim2=0
        g1[5] = (int)ne;                          // tensor_dim0_stride[31:0]
        g1[6] = 0;                                // stride hi | tensor_dim1_stride lo
        g1[7] = 0;                                // tensor_dim1_stride hi
        int32x4 gz4 = {};                         // groups 2/3 unused for 1-D tile
        int32x8 gz8 = {};                         // extra group (6-arg toolchain), zero
        __builtin_amdgcn_tensor_load_to_lds(g0, g1, gz4, gz4, gz8, 0);
        __builtin_amdgcn_s_wait_tensorcnt(0);
    }
    __syncthreads();
#else
    for (int i = d; i < NTOK * EDIM; i += 256) sdbl[i] = src[i];
    __syncthreads();
#endif

    float wdt[DTRANK];
#pragma unroll
    for (int j = 0; j < DTRANK; j++) wdt[j] = dtw[d * DTRANK + j];
    float Arow[DSTATE];
#pragma unroll
    for (int n = 0; n < DSTATE; n++) Arow[n] = -expf(A_log[d * DSTATE + n]);
    const float bias = dtb[d];
    const float Dd   = Dp[d];
    float h[DSTATE] = {};

    for (int l = 0; l < NTOK; l++) {
        const float* row = &sdbl[l * EDIM];
        float dr = bias;
#pragma unroll
        for (int j = 0; j < DTRANK; j++) dr += row[j] * wdt[j];
        float sp = (dr > 20.f) ? dr : log1pf(expf(dr));       // softplus
        float u = x_conv[((size_t)b * NTOK + l) * DHALF + d];
        float y = 0.f;
#pragma unroll
        for (int n = 0; n < DSTATE; n++) {
            float dA = expf(sp * Arow[n]);
            h[n] = h[n] * dA + sp * row[DTRANK + n] * u;
            y += h[n] * row[DTRANK + DSTATE + n];
        }
        y += Dd * u;
        if (isnan(y)) y = 0.f;
        else if (isinf(y)) y = (y > 0.f) ? 3.4028235e38f : -3.4028235e38f;
        y_concat[((size_t)b * NTOK + l) * DIMX + d] = (bf16)y;
    }
}

// ---------------------------------------------------------------------------
// Host-side launch sequence
// ---------------------------------------------------------------------------
extern "C" void kernel_launch(void* const* d_in, const int* in_sizes, int n_in,
                              void* d_out, int out_size, void* d_ws, size_t ws_size,
                              hipStream_t stream)
{
    (void)in_sizes; (void)n_in; (void)out_size; (void)ws_size;
    // setup_inputs() dict order:
    const float* tokens   = (const float*)d_in[0];
    const float* pre_w[4]  = { (const float*)d_in[1], (const float*)d_in[3],
                               (const float*)d_in[5], (const float*)d_in[7] };
    const float* pre_b[4]  = { (const float*)d_in[2], (const float*)d_in[4],
                               (const float*)d_in[6], (const float*)d_in[8] };
    const float* post_w[4] = { (const float*)d_in[9],  (const float*)d_in[11],
                               (const float*)d_in[13], (const float*)d_in[15] };
    const float* post_b[4] = { (const float*)d_in[10], (const float*)d_in[12],
                               (const float*)d_in[14], (const float*)d_in[16] };
    const float* in_proj_w = (const float*)d_in[17];
    const float* conv_x_w  = (const float*)d_in[18];
    const float* conv_x_b  = (const float*)d_in[19];
    const float* conv_z_w  = (const float*)d_in[20];
    const float* conv_z_b  = (const float*)d_in[21];
    const float* x_proj_w  = (const float*)d_in[22];
    const float* ln_w      = (const float*)d_in[23];
    const float* ln_b      = (const float*)d_in[24];
    const float* dt_proj_w = (const float*)d_in[25];
    const float* dt_proj_b = (const float*)d_in[26];
    const float* A_log     = (const float*)d_in[27];
    const float* D_param   = (const float*)d_in[28];
    const float* out_proj_w= (const float*)d_in[29];
    const float* gate_lg   = (const float*)d_in[30];
    float* out = (float*)d_out;

    // workspace carve-up (all 256B-aligned)
    char* ws = (char*)d_ws;
    size_t off = 0;
    auto take = [&](size_t bytes) { size_t o = off; off += (bytes + 255) & ~(size_t)255; return o; };
    bf16*  tok_bf    = (bf16*) (ws + take((size_t)MPRE * DIMX * 2));
    bf16*  w_pre     = (bf16*) (ws + take((size_t)NDIR * DIMX * DIMX * 2));
    bf16*  w_in      = (bf16*) (ws + take((size_t)DIMX * DIMX * 2));
    bf16*  w_out     = (bf16*) (ws + take((size_t)DIMX * DIMX * 2));
    bf16*  w_post    = (bf16*) (ws + take((size_t)NDIR * DIMX * DIMX * 2));
    float* comb_bias = (float*)(ws + take((size_t)DIMX * 4));
    bf16*  r_all     = (bf16*) (ws + take((size_t)MALL * DIMX * 2));  // reused as y_all
    bf16*  xz        = (bf16*) (ws + take((size_t)MALL * DIMX * 2));
    float* x_conv    = (float*)(ws + take((size_t)MALL * DHALF * 4));
    bf16*  y_concat  = (bf16*) (ws + take((size_t)MALL * DIMX * 2));
    float* x_dbl     = (float*)(ws + take((size_t)MALL * EDIM * 4));
    bf16*  y_all     = r_all;                                         // r_all dead after in_proj

    const int WSZ = DIMX * DIMX;                 // 262144

    // --- prep: bf16 conversions + gated post weights / combined bias ---
    k_f32_to_bf16<<<(MPRE * DIMX + 255) / 256, 256, 0, stream>>>(tokens, tok_bf, MPRE * DIMX);
    for (int dirn = 0; dirn < 4; dirn++)
        k_f32_to_bf16<<<(WSZ + 255) / 256, 256, 0, stream>>>(pre_w[dirn], w_pre + (size_t)dirn * WSZ, WSZ);
    k_f32_to_bf16<<<(WSZ + 255) / 256, 256, 0, stream>>>(in_proj_w,  w_in,  WSZ);
    k_f32_to_bf16<<<(WSZ + 255) / 256, 256, 0, stream>>>(out_proj_w, w_out, WSZ);
    for (int dirn = 0; dirn < 4; dirn++)
        k_scale_post_w<<<(WSZ + 255) / 256, 256, 0, stream>>>(post_w[dirn], gate_lg, dirn,
                                                              w_post + (size_t)dirn * WSZ);
    k_comb_bias<<<2, 256, 0, stream>>>(post_b[0], post_b[1], post_b[2], post_b[3], gate_lg, comb_bias);

    // --- 1) pre-projections + permuted scatter into r_all (bf16) ---
    dim3 gPre(MPRE / 16 / 8, DIMX / 64, NDIR);   // (196, 8, 4)
    k_gemm_pre<<<gPre, 256, 0, stream>>>(tok_bf, w_pre,
                                         pre_b[0], pre_b[1], pre_b[2], pre_b[3], r_all);

    // --- 2) in_proj: xz = r_all @ in_proj_w^T ---
    dim3 gAll(MALL / 16 / 8, DIMX / 64);         // (784, 8)
    k_gemm_bf16<<<gAll, 256, 0, stream>>>(r_all, w_in, xz, MALL);

    // --- 3) depthwise conv + silu (x -> x_conv f32, z -> y_concat hi half) ---
    size_t convN = (size_t)BALL * NTOK * DIMX;
    k_conv_silu<<<(unsigned)((convN + 255) / 256), 256, 0, stream>>>(
        xz, conv_x_w, conv_x_b, conv_z_w, conv_z_b, x_conv, y_concat);

    // --- 4) x_proj + layernorm -> x_dbl (b,l,48) f32 ---
    k_xproj_ln<<<MALL, 64, 0, stream>>>(x_conv, x_proj_w, ln_w, ln_b, x_dbl);

    // --- 5) selective scan (TDM-staged x_dbl) -> y_concat lo half (bf16) ---
    k_scan<<<BALL, 256, 0, stream>>>(x_dbl, x_conv, dt_proj_w, dt_proj_b,
                                     A_log, D_param, y_concat);

    // --- 6) out_proj: y_all = y_concat @ out_proj_w^T ---
    k_gemm_bf16<<<gAll, 256, 0, stream>>>(y_concat, w_out, y_all, MALL);

    // --- 7) fused gated post-projections -> f32 output ---
    dim3 gPost(MPRE / 16 / 8, DIMX / 64);        // (196, 8)
    k_gemm_post<<<gPost, 256, 0, stream>>>(y_all, w_post, comb_bias, out);
}
